// MeanAggSelfAttentionBlock_59682865545488
// MI455X (gfx1250) — matrix operand order repaired and checked by reading, not compiled, not run on hardware
//
#include <hip/hip_runtime.h>
#include <hip/hip_bf16.h>

// ---------------- problem constants ----------------
#define BB 16
#define NN 1024
#define DD 512
#define HH 8
#define HD 64
#define LL 4
#define DF 2048          // WF * D
#define MM (BB * NN)     // 16384 rows

typedef __attribute__((ext_vector_type(16))) _Float16 v16h;
typedef __attribute__((ext_vector_type(8)))  _Float16 h8;
typedef __attribute__((ext_vector_type(8)))  float    v8f;

// ---------------- CDNA5 async global->LDS copy ----------------
// addrspace(3) pointers cast to flat keep the LDS byte offset in the low
// 32 bits, which is exactly the VDST (LDS address) operand of the async op.
__device__ inline void async_ld_b128(void* lds_dst, const void* gsrc) {
  unsigned lds = (unsigned)(unsigned long long)(uintptr_t)lds_dst;
  unsigned long long ga = (unsigned long long)(uintptr_t)gsrc;
  asm volatile("global_load_async_to_lds_b128 %0, %1, off"
               :: "v"(lds), "v"(ga) : "memory");
}
__device__ inline void wait_async0() {
  asm volatile("s_wait_asynccnt 0x0" ::: "memory");
}
__device__ inline void wait_ds0() {
  asm volatile("s_wait_dscnt 0x0" ::: "memory");
}

// ---------------- fragment helpers ----------------
// A-fragment (16x32 f16): lanes 0-15 -> rows 0-15, K=0..15;
// lanes 16-31 -> rows 0-15, K=16..31.
__device__ inline v16h load_a_frag(const _Float16* base, int stride, int lane) {
  int row = lane & 15;
  int kh  = (lane >> 4) * 16;
  const _Float16* p = base + row * stride + kh;
  h8 lo = *(const h8*)(p);
  h8 hi = *(const h8*)(p + 8);
  v16h a;
#pragma unroll
  for (int i = 0; i < 8; ++i) { a[i] = lo[i]; a[i + 8] = hi[i]; }
  return a;
}

// B-fragment (32x16 f16) from a [col][k] buffer: lanes 0-15 -> cols 0-15
// covering K=0..15; lanes 16-31 same cols covering K=16..31.
__device__ inline v16h load_b_fragT(const _Float16* baseT, int stride, int lane) {
  int col = lane & 15;
  int k0  = (lane >> 4) * 16;
  const _Float16* p = baseT + col * stride + k0;
  h8 lo = *(const h8*)(p);
  h8 hi = *(const h8*)(p + 8);
  v16h b;
#pragma unroll
  for (int i = 0; i < 8; ++i) { b[i] = lo[i]; b[i + 8] = hi[i]; }
  return b;
}

__device__ inline v8f wmma16(v16h a, v16h b, v8f c) {
  return __builtin_amdgcn_wmma_f32_16x16x32_f16(false, a, false, b, (short)0, c,
                                                false, false);
}

// ---------------- f32 -> f16 weight convert + transpose ----------------
// in : w [L][K][Nc] f32 ; out: wT [L][Nc][K] f16
__global__ void cvt_tr_kernel(const float* __restrict__ s,
                              _Float16* __restrict__ d, int K, int Nc) {
  int l = blockIdx.y;
  int i = blockIdx.x * 256 + threadIdx.x;
  if (i < K * Nc) {
    int k = i / Nc, n = i - k * Nc;
    d[(size_t)l * K * Nc + (size_t)n * K + k] =
        (_Float16)s[(size_t)l * K * Nc + i];
  }
}

// ---------------- LayerNorm -> f16 ----------------
__global__ void layernorm_f16_kernel(const float* __restrict__ h,
                                     const float* __restrict__ w,
                                     const float* __restrict__ bsrc,
                                     _Float16* __restrict__ y16) {
  int row = blockIdx.x;
  int t   = threadIdx.x;  // 256 threads, 2 elems each
  __shared__ float red[256];
  const float* hr = h + (size_t)row * DD;
  float x0 = hr[t], x1 = hr[t + 256];
  red[t] = x0 + x1;
  __syncthreads();
  for (int s = 128; s > 0; s >>= 1) {
    if (t < s) red[t] += red[t + s];
    __syncthreads();
  }
  float mu = red[0] * (1.0f / DD);
  __syncthreads();
  float d0 = x0 - mu, d1 = x1 - mu;
  red[t] = d0 * d0 + d1 * d1;
  __syncthreads();
  for (int s = 128; s > 0; s >>= 1) {
    if (t < s) red[t] += red[t + s];
    __syncthreads();
  }
  float rs = rsqrtf(red[0] * (1.0f / DD) + 1e-5f);
  _Float16* yr = y16 + (size_t)row * DD;
  yr[t]       = (_Float16)(d0 * rs * w[t] + bsrc[t]);
  yr[t + 256] = (_Float16)(d1 * rs * w[t + 256] + bsrc[t + 256]);
}

// ---------------- tiled WMMA GEMM (double-buffered async staging) ----------
// C[M x Nc] = A[M x K](f16) * BwT[Nc x K](f16, pre-transposed) + bias
// modes: 0 f16 out; 1 f16 gelu out; 2 f32 out + residual
#define GM 128
#define GN 128
#define GK 32
#define ASTR 48
__global__ __launch_bounds__(256) void gemm_wmma_kernel(
    const _Float16* __restrict__ A, const _Float16* __restrict__ BwT,
    const float* __restrict__ bias, _Float16* __restrict__ out16,
    float* __restrict__ outF, const float* __restrict__ res, int M, int K,
    int Nc, int mode) {
  __shared__ __align__(16) _Float16 As[2][GM * ASTR];
  __shared__ __align__(16) _Float16 BsT[2][GN * ASTR];

  int tid  = threadIdx.x;
  int lane = tid & 31;
  int wave = tid >> 5;          // 8 waves
  int wm = wave >> 1;           // 4 row-waves (32 rows each)
  int wn = wave & 1;            // 2 col-waves (64 cols each)
  int rowBlock = blockIdx.y * GM;
  int colBlock = blockIdx.x * GN;

  v8f acc[2][4] = {};

  // stage one K-tile (buf) via async global->LDS
  auto stage = [&](int kb, int buf) {
#pragma unroll
    for (int j = 0; j < 2; ++j) {
      int s = tid + j * 256;            // 512 chunks of 8 halves
      int r = s >> 2, c = (s & 3) * 8;  // row / half-chunk
      async_ld_b128(&As[buf][r * ASTR + c],
                    &A[(size_t)(rowBlock + r) * K + kb + c]);
      async_ld_b128(&BsT[buf][r * ASTR + c],
                    &BwT[(size_t)(colBlock + r) * K + kb + c]);
    }
  };

  stage(0, 0);
  wait_async0();
  __syncthreads();

  int nk = K / GK;
  for (int t = 0; t < nk; ++t) {
    int cur = t & 1;
    if (t + 1 < nk) stage((t + 1) * GK, cur ^ 1);

    v16h af[2], bf[4];
    af[0] = load_a_frag(&As[cur][(wm * 32) * ASTR], ASTR, lane);
    af[1] = load_a_frag(&As[cur][(wm * 32 + 16) * ASTR], ASTR, lane);
#pragma unroll
    for (int nt = 0; nt < 4; ++nt)
      bf[nt] = load_b_fragT(&BsT[cur][(wn * 64 + nt * 16) * ASTR], ASTR, lane);
#pragma unroll
    for (int mt = 0; mt < 2; ++mt)
#pragma unroll
      for (int nt = 0; nt < 4; ++nt)
        acc[mt][nt] = wmma16(af[mt], bf[nt], acc[mt][nt]);

    wait_ds0();      // our ds_loads done before next overwrite
    wait_async0();   // next tile landed in LDS
    __syncthreads();
  }

  // epilogue: lane 0-15 -> col=lane, rows r; lane 16-31 -> col=lane-16, rows 8+r
  int lr = (lane >> 4) * 8;
  int lc = lane & 15;
#pragma unroll
  for (int mt = 0; mt < 2; ++mt) {
#pragma unroll
    for (int nt = 0; nt < 4; ++nt) {
      int gc = colBlock + wn * 64 + nt * 16 + lc;
      float bb = bias[gc];
      int gr0 = rowBlock + wm * 32 + mt * 16 + lr;
#pragma unroll
      for (int r = 0; r < 8; ++r) {
        float v = acc[mt][nt][r] + bb;
        size_t idx = (size_t)(gr0 + r) * Nc + gc;
        if (mode == 2) {
          outF[idx] = res[idx] + v;
        } else {
          if (mode == 1) v = 0.5f * v * (1.0f + erff(v * 0.70710678f));
          out16[idx] = (_Float16)v;
        }
      }
    }
  }
}

// ---------------- RoPE + head pack  [B,N,H,HD] -> [B*H][N][HD] ----------------
__global__ void rope_pack_kernel(const _Float16* __restrict__ src,
                                 _Float16* __restrict__ dst, int do_rope,
                                 float scale) {
  unsigned p = blockIdx.x * 256 + threadIdx.x;  // pair index
  int hd2 = p & (HD / 2 - 1);
  unsigned rest = p / (HD / 2);
  int hh = rest & (HH - 1);
  rest /= HH;
  int n = rest & (NN - 1);
  unsigned b = rest / NN;
  size_t si = ((((size_t)b * NN + n) * HH + hh) * HD) + hd2 * 2;
  float a = (float)src[si], c = (float)src[si + 1];
  if (do_rope) {
    float freq = __powf(10000.0f, -(float)(2 * hd2) / (float)HD);
    float ang = (float)n * freq;
    float cs = __cosf(ang), sn = __sinf(ang);
    float na = a * cs - c * sn;
    float nc = c * cs + a * sn;
    a = na; c = nc;
  }
  a *= scale; c *= scale;
  size_t di = ((((size_t)b * HH + hh) * NN + n) * HD) + hd2 * 2;
  dst[di] = (_Float16)a;
  dst[di + 1] = (_Float16)c;
}

// ---------------- flash attention (WMMA, async V staging) ----------------
// grid (N/64, B*H), block 128 (4 waves). Each wave owns 16 query rows.
#define VSTR 72
__global__ __launch_bounds__(128) void flash_attn_kernel(
    const _Float16* __restrict__ Qp, const _Float16* __restrict__ Kp,
    const _Float16* __restrict__ Vp, _Float16* __restrict__ o16) {
  __shared__ __align__(16) _Float16 Vs[32 * VSTR];     // [key][hd] for key block
  __shared__ __align__(16) _Float16 Pb[4][16 * ASTR];  // per-wave P tile [row][key]

  int tid = threadIdx.x, lane = tid & 31, wave = tid >> 5;
  int bh = blockIdx.y;
  int b = bh / HH, hh = bh - b * HH;
  int qbase = blockIdx.x * 64 + wave * 16;
  const _Float16* Q  = Qp + (size_t)bh * NN * HD;
  const _Float16* Kk = Kp + (size_t)bh * NN * HD;
  const _Float16* Vv = Vp + (size_t)bh * NN * HD;

  // Q fragments (HD=64 -> two 16x32 A-fragments), loaded once
  v16h aq[2];
  {
    int row = qbase + (lane & 15);
    int kh = (lane >> 4) * 16;
#pragma unroll
    for (int f = 0; f < 2; ++f) {
      const _Float16* p = Q + (size_t)row * HD + f * 32 + kh;
      h8 lo = *(const h8*)p;
      h8 hi = *(const h8*)(p + 8);
#pragma unroll
      for (int i = 0; i < 8; ++i) { aq[f][i] = lo[i]; aq[f][i + 8] = hi[i]; }
    }
  }

  v8f accO[4] = {};
  float mrun[8], lrun[8];
#pragma unroll
  for (int r = 0; r < 8; ++r) { mrun[r] = -3.0e38f; lrun[r] = 0.0f; }

  for (int kb = 0; kb < NN; kb += 32) {
    wait_ds0();
    __syncthreads();
    // async stage V block [32 keys][64 hd]
#pragma unroll
    for (int j = 0; j < 2; ++j) {
      int s = tid + j * 128;
      int key = s >> 3, c = (s & 7) * 8;
      async_ld_b128(&Vs[key * VSTR + c], &Vv[(size_t)(kb + key) * HD + c]);
    }
    wait_async0();
    __syncthreads();

    // scores: two 16x16 tiles (key cols 0-15 / 16-31), contraction HD=64
    v8f s0 = {}, s1 = {};
#pragma unroll
    for (int f = 0; f < 2; ++f) {
      v16h bk0, bk1;
      {
        const _Float16* p =
            Kk + (size_t)(kb + (lane & 15)) * HD + f * 32 + (lane >> 4) * 16;
        h8 lo = *(const h8*)p; h8 hi = *(const h8*)(p + 8);
#pragma unroll
        for (int i = 0; i < 8; ++i) { bk0[i] = lo[i]; bk0[i + 8] = hi[i]; }
      }
      {
        const _Float16* p =
            Kk + (size_t)(kb + 16 + (lane & 15)) * HD + f * 32 + (lane >> 4) * 16;
        h8 lo = *(const h8*)p; h8 hi = *(const h8*)(p + 8);
#pragma unroll
        for (int i = 0; i < 8; ++i) { bk1[i] = lo[i]; bk1[i + 8] = hi[i]; }
      }
      s0 = wmma16(aq[f], bk0, s0);
      s1 = wmma16(aq[f], bk1, s1);
    }

    // online softmax over this 32-key block
    float p0[8], p1[8], alpha[8];
#pragma unroll
    for (int r = 0; r < 8; ++r) {
      float rm = fmaxf(s0[r], s1[r]);
#pragma unroll
      for (int off = 1; off < 16; off <<= 1)
        rm = fmaxf(rm, __shfl_xor(rm, off, 32));
      float mnew = fmaxf(mrun[r], rm);
      alpha[r] = __expf(mrun[r] - mnew);
      mrun[r] = mnew;
      p0[r] = __expf(s0[r] - mnew);
      p1[r] = __expf(s1[r] - mnew);
      float rs = p0[r] + p1[r];
#pragma unroll
      for (int off = 1; off < 16; off <<= 1) rs += __shfl_xor(rs, off, 32);
      lrun[r] = lrun[r] * alpha[r] + rs;
    }
#pragma unroll
    for (int nt = 0; nt < 4; ++nt)
#pragma unroll
      for (int r = 0; r < 8; ++r) accO[nt][r] *= alpha[r];

    // P (16x32) -> per-wave LDS, reload in A-fragment layout
    _Float16* pb = &Pb[wave][0];
    {
      int rrow = (lane >> 4) * 8;
      int cc = lane & 15;
#pragma unroll
      for (int r = 0; r < 8; ++r) {
        pb[(rrow + r) * ASTR + cc] = (_Float16)p0[r];
        pb[(rrow + r) * ASTR + 16 + cc] = (_Float16)p1[r];
      }
    }
    wait_ds0();
    v16h ap = load_a_frag(pb, ASTR, lane);

    // O += P @ V  (4 hd-tiles of 16); B-frag gathered from Vs[key][hd]
#pragma unroll
    for (int nt = 0; nt < 4; ++nt) {
      v16h bv;
      int hd = nt * 16 + (lane & 15);
      int k0 = (lane >> 4) * 16;
#pragma unroll
      for (int e = 0; e < 16; ++e) bv[e] = Vs[(k0 + e) * VSTR + hd];
      accO[nt] = wmma16(ap, bv, accO[nt]);
    }
  }

  // write O to [B,N,D] f16 for the output projection
  int lc = lane & 15;
  int lr = (lane >> 4) * 8;
#pragma unroll
  for (int r = 0; r < 8; ++r) {
    float inv = 1.0f / lrun[r];
    int nrow = qbase + lr + r;
#pragma unroll
    for (int nt = 0; nt < 4; ++nt) {
      float o = accO[nt][r] * inv;
      o16[((size_t)(b * NN + nrow)) * DD + hh * HD + nt * 16 + lc] = (_Float16)o;
    }
  }
}

// ---------------- mean pool over N ----------------
__global__ void pool_kernel(const float* __restrict__ h, float* __restrict__ out) {
  int b = blockIdx.x;
  int d = threadIdx.x;  // 512 threads
  float s = 0.0f;
  for (int n = 0; n < NN; ++n) s += h[((size_t)b * NN + n) * DD + d];
  out[(size_t)b * DD + d] = s * (1.0f / NN);
}

// ---------------- host orchestration ----------------
extern "C" void kernel_launch(void* const* d_in, const int* in_sizes, int n_in,
                              void* d_out, int out_size, void* d_ws,
                              size_t ws_size, hipStream_t stream) {
  const float* x    = (const float*)d_in[0];
  const float* ln1w = (const float*)d_in[2];
  const float* ln1b = (const float*)d_in[3];
  const float* wq   = (const float*)d_in[4];
  const float* bq   = (const float*)d_in[5];
  const float* wk   = (const float*)d_in[6];
  const float* bk   = (const float*)d_in[7];
  const float* wv   = (const float*)d_in[8];
  const float* bv   = (const float*)d_in[9];
  const float* wo   = (const float*)d_in[10];
  const float* bo   = (const float*)d_in[11];
  const float* ln2w = (const float*)d_in[12];
  const float* ln2b = (const float*)d_in[13];
  const float* w1   = (const float*)d_in[14];
  const float* b1   = (const float*)d_in[15];
  const float* w2   = (const float*)d_in[16];
  const float* b2   = (const float*)d_in[17];

  char* ws = (char*)d_ws;
  size_t off = 0;
  auto carve = [&](size_t bytes) {
    void* p = ws + off;
    off += (bytes + 255) & ~(size_t)255;
    return p;
  };
  float*    h    = (float*)carve((size_t)MM * DD * 4);
  _Float16* y16  = (_Float16*)carve((size_t)MM * DD * 2);
  _Float16* qraw = (_Float16*)carve((size_t)MM * DD * 2);
  _Float16* kraw = (_Float16*)carve((size_t)MM * DD * 2);
  _Float16* vraw = (_Float16*)carve((size_t)MM * DD * 2);
  _Float16* qp   = (_Float16*)carve((size_t)MM * DD * 2);
  _Float16* kp   = (_Float16*)carve((size_t)MM * DD * 2);
  _Float16* vp   = (_Float16*)carve((size_t)MM * DD * 2);
  _Float16* o16  = (_Float16*)carve((size_t)MM * DD * 2);
  _Float16* m16  = (_Float16*)carve((size_t)MM * DF * 2);
  _Float16* wq16 = (_Float16*)carve((size_t)LL * DD * DD * 2);
  _Float16* wk16 = (_Float16*)carve((size_t)LL * DD * DD * 2);
  _Float16* wv16 = (_Float16*)carve((size_t)LL * DD * DD * 2);
  _Float16* wo16 = (_Float16*)carve((size_t)LL * DD * DD * 2);
  _Float16* w116 = (_Float16*)carve((size_t)LL * DD * DF * 2);
  _Float16* w216 = (_Float16*)carve((size_t)LL * DF * DD * 2);

  // h = x
  hipMemcpyAsync(h, x, (size_t)MM * DD * 4, hipMemcpyDeviceToDevice, stream);

  // weights -> f16 transposed [L][Nc][K]
  auto cvt = [&](const float* s, _Float16* d, int K, int Nc) {
    cvt_tr_kernel<<<dim3((K * Nc + 255) / 256, LL), 256, 0, stream>>>(s, d, K, Nc);
  };
  cvt(wq, wq16, DD, DD);
  cvt(wk, wk16, DD, DD);
  cvt(wv, wv16, DD, DD);
  cvt(wo, wo16, DD, DD);
  cvt(w1, w116, DD, DF);
  cvt(w2, w216, DF, DD);

  const int ropeBlocks = MM * DD / 2 / 256;

  for (int l = 0; l < LL; ++l) {
    layernorm_f16_kernel<<<MM, 256, 0, stream>>>(h, ln1w + l * DD, ln1b + l * DD, y16);

    dim3 gqkv(DD / GN, MM / GM);
    gemm_wmma_kernel<<<gqkv, 256, 0, stream>>>(y16, wq16 + (size_t)l * DD * DD,
                                               bq + l * DD, qraw, nullptr, nullptr,
                                               MM, DD, DD, 0);
    gemm_wmma_kernel<<<gqkv, 256, 0, stream>>>(y16, wk16 + (size_t)l * DD * DD,
                                               bk + l * DD, kraw, nullptr, nullptr,
                                               MM, DD, DD, 0);
    gemm_wmma_kernel<<<gqkv, 256, 0, stream>>>(y16, wv16 + (size_t)l * DD * DD,
                                               bv + l * DD, vraw, nullptr, nullptr,
                                               MM, DD, DD, 0);

    rope_pack_kernel<<<ropeBlocks, 256, 0, stream>>>(qraw, qp, l == 0 ? 1 : 0, 0.125f);
    rope_pack_kernel<<<ropeBlocks, 256, 0, stream>>>(kraw, kp, l == 0 ? 1 : 0, 1.0f);
    rope_pack_kernel<<<ropeBlocks, 256, 0, stream>>>(vraw, vp, 0, 1.0f);

    flash_attn_kernel<<<dim3(NN / 64, BB * HH), 128, 0, stream>>>(qp, kp, vp, o16);

    gemm_wmma_kernel<<<gqkv, 256, 0, stream>>>(o16, wo16 + (size_t)l * DD * DD,
                                               bo + l * DD, nullptr, h, h,
                                               MM, DD, DD, 2);

    layernorm_f16_kernel<<<MM, 256, 0, stream>>>(h, ln2w + l * DD, ln2b + l * DD, y16);

    gemm_wmma_kernel<<<dim3(DF / GN, MM / GM), 256, 0, stream>>>(
        y16, w116 + (size_t)l * DD * DF, b1 + l * DF, m16, nullptr, nullptr,
        MM, DD, DF, 1);
    gemm_wmma_kernel<<<gqkv, 256, 0, stream>>>(m16, w216 + (size_t)l * DF * DD,
                                               b2 + l * DD, nullptr, h, h,
                                               MM, DF, DD, 2);
  }

  pool_kernel<<<BB, DD, 0, stream>>>(h, (float*)d_out);
}
